// GroupedQueryAttention_77790447665561
// MI455X (gfx1250) — matrix operand order repaired and checked by reading, not compiled, not run on hardware
//
#include <hip/hip_runtime.h>
#include <hip/hip_bf16.h>

#define USE_ASYNC 1   // gfx1250 GLOBAL_LOAD_ASYNC_TO_LDS path (ASYNCcnt-tracked)

typedef __bf16 bf16;
typedef __attribute__((ext_vector_type(16))) __bf16 bf16x16;
typedef __attribute__((ext_vector_type(8)))  float  f32x8;

// ---------------------------------------------------------------------------
// Async global -> LDS staging (CDNA5). One b128 per lane per issue; tracked by
// ASYNCcnt. Fallback: plain load/store through VGPRs.
// ---------------------------------------------------------------------------
__device__ __forceinline__ void stage_b128(void* lds_p, const void* gp) {
#if USE_ASYNC
  const unsigned la =
      (unsigned)(unsigned long long)(__attribute__((address_space(3))) void*)lds_p;
  asm volatile("global_load_async_to_lds_b128 %0, %1, off"
               :
               : "v"(la), "v"(gp)
               : "memory");
#else
  *(uint4*)lds_p = *(const uint4*)gp;
#endif
}

__device__ __forceinline__ void stage_wait() {
#if USE_ASYNC
  asm volatile("s_wait_asynccnt 0" ::: "memory");
#endif
}

// ---------------------------------------------------------------------------
// WMMA fragment loaders, wave32 (CDNA5 ISA 7.12.2).
// A (16x32 row-major tile): lane=half*16+row ->
//   a[0..7]  = A[row][kofs+half*8 .. +7]       (16B run)
//   a[8..15] = A[row][kofs+16+half*8 .. +7]    (16B run)
// B (32x16) from N-MAJOR tile Bt[n][k]: lane=half*16+col ->
//   b[0..7]  = Bt[col][kofs+half*8 .. +7]      (16B run)
//   b[8..15] = Bt[col][kofs+16+half*8 .. +7]   (16B run)
// C/D (16x16 f32): VGPR r -> row M = half*8+r, col = lane&15
// ---------------------------------------------------------------------------
__device__ __forceinline__ bf16x16 ld_frag_pair(const bf16* p0, const bf16* p1) {
  union { uint4 u[2]; bf16x16 v; } f;
  f.u[0] = *(const uint4*)p0;   // ds_load_b128
  f.u[1] = *(const uint4*)p1;   // ds_load_b128
  return f.v;
}

__device__ __forceinline__ bf16x16 a_frag(const bf16* tile, int ld, int lane,
                                          int kofs) {
  const int half = lane >> 4, row = lane & 15;
  const bf16* p = tile + row * ld + kofs + half * 8;
  return ld_frag_pair(p, p + 16);
}

__device__ __forceinline__ bf16x16 bn_frag(const bf16* tile, int ld, int lane,
                                           int kofs) {
  const int half = lane >> 4, col = lane & 15;
  const bf16* p = tile + col * ld + kofs + half * 8;
  return ld_frag_pair(p, p + 16);
}

__device__ __forceinline__ f32x8 wmma_bf16(bf16x16 a, bf16x16 b, f32x8 c) {
  return __builtin_amdgcn_wmma_f32_16x16x32_bf16(false, a, false, b, (short)0, c,
                                                 false, false);
}

// ---------------------------------------------------------------------------
// f32 -> bf16 convert (4 elems / thread; n is a multiple of 1024)
// ---------------------------------------------------------------------------
__global__ void cvt_f32_bf16(const float* __restrict__ in, bf16* __restrict__ out,
                             int n) {
  int i = (blockIdx.x * blockDim.x + threadIdx.x) * 4;
  if (i + 3 < n) {
    float4 f = *(const float4*)(in + i);
    out[i]     = (bf16)f.x;
    out[i + 1] = (bf16)f.y;
    out[i + 2] = (bf16)f.z;
    out[i + 3] = (bf16)f.w;
  }
}

// ---------------------------------------------------------------------------
// Fused convert + transpose: W[K][N] f32 -> Wt[N][K] bf16. 32x32 LDS tile.
// ---------------------------------------------------------------------------
__global__ __launch_bounds__(256) void cvt_t_f32_bf16(const float* __restrict__ in,
                                                      bf16* __restrict__ out,
                                                      int K, int N) {
  __shared__ bf16 t[32][33];
  const int tid = threadIdx.x;
  const int n0 = blockIdx.x * 32, k0 = blockIdx.y * 32;
#pragma unroll
  for (int i = 0; i < 4; ++i) {
    const int idx = tid + i * 256;
    const int kk = idx >> 5, nn = idx & 31;
    t[kk][nn] = (bf16)in[(size_t)(k0 + kk) * N + n0 + nn];
  }
  __syncthreads();
#pragma unroll
  for (int i = 0; i < 4; ++i) {
    const int idx = tid + i * 256;
    const int nn = idx >> 5, kk = idx & 31;
    out[(size_t)(n0 + nn) * K + k0 + kk] = t[kk][nn];
  }
}

// ---------------------------------------------------------------------------
// Batched bf16 transpose: per z: in[R][C] -> out[C][R]. grid=(C/32, R/32, B)
// ---------------------------------------------------------------------------
__global__ __launch_bounds__(256) void transpose_bf16(const bf16* __restrict__ in,
                                                      bf16* __restrict__ out,
                                                      int R, int C) {
  __shared__ bf16 t[32][33];
  const size_t base = (size_t)blockIdx.z * R * C;
  const int tid = threadIdx.x;
  const int c0 = blockIdx.x * 32, r0 = blockIdx.y * 32;
#pragma unroll
  for (int i = 0; i < 4; ++i) {
    const int idx = tid + i * 256;
    const int rr = idx >> 5, cc = idx & 31;
    t[rr][cc] = in[base + (size_t)(r0 + rr) * C + c0 + cc];
  }
  __syncthreads();
#pragma unroll
  for (int i = 0; i < 4; ++i) {
    const int idx = tid + i * 256;
    const int rr = idx >> 5, cc = idx & 31;
    out[base + (size_t)(c0 + rr) * R + r0 + cc] = t[cc][rr];
  }
}

// ---------------------------------------------------------------------------
// GEMM: C[M,N] = A[M,K](bf16) @ Bt[N,K](bf16, pre-transposed) + bias[N](f32)
// Block 128x128, 8 waves (wave tile 32x64, 2x4 accums). k-step 32 with
// double-buffered LDS filled by async global->LDS; one barrier per k-step.
// M%128==0, N%128==0, K%32==0.
// ---------------------------------------------------------------------------
template <bool BF16OUT>
__global__ __launch_bounds__(256) void gemm_bias_kernel(
    const bf16* __restrict__ A, const bf16* __restrict__ Bt,
    const float* __restrict__ bias, void* __restrict__ Cout, int M, int N, int K) {
  __shared__ bf16 As[2][128][40];   // ping-pong 128x32, pad->40 (80B rows)
  __shared__ bf16 Bs[2][128][40];   // n-major 128x32

  const int tid = threadIdx.x, lane = tid & 31, wave = tid >> 5;
  const int m0 = blockIdx.x * 128, n0 = blockIdx.y * 128;
  const int wm = (wave & 3) * 32, wn = (wave >> 2) * 64;

  // staging geometry: 512 16B-chunks per tile -> 2 per thread
  int rr[2], cc[2];
  size_t aoff[2], boff[2];
#pragma unroll
  for (int c = 0; c < 2; ++c) {
    const int idx = tid + c * 256;
    rr[c] = idx >> 2;             // 0..127
    cc[c] = (idx & 3) * 8;        // 0,8,16,24
    aoff[c] = (size_t)(m0 + rr[c]) * K + cc[c];
    boff[c] = (size_t)(n0 + rr[c]) * K + cc[c];
  }

  // prologue: fill buffer 0
#pragma unroll
  for (int c = 0; c < 2; ++c) {
    stage_b128(&As[0][rr[c]][cc[c]], A + aoff[c]);
    stage_b128(&Bs[0][rr[c]][cc[c]], Bt + boff[c]);
  }
  stage_wait();
  __syncthreads();

  f32x8 acc[2][4] = {};
  int cur = 0;
  for (int k0 = 0; k0 < K; k0 += 32) {
    const bool pre = (k0 + 32) < K;
    if (pre) {  // async-fill the other buffer while we compute
      const int nxt = cur ^ 1;
#pragma unroll
      for (int c = 0; c < 2; ++c) {
        stage_b128(&As[nxt][rr[c]][cc[c]], A + aoff[c] + k0 + 32);
        stage_b128(&Bs[nxt][rr[c]][cc[c]], Bt + boff[c] + k0 + 32);
      }
    }
    const bf16* Ac = &As[cur][0][0];
    const bf16* Bc = &Bs[cur][0][0];
    bf16x16 af[2], bfr[4];
    af[0] = a_frag(Ac + wm * 40, 40, lane, 0);
    af[1] = a_frag(Ac + (wm + 16) * 40, 40, lane, 0);
#pragma unroll
    for (int j = 0; j < 4; ++j)
      bfr[j] = bn_frag(Bc + (wn + j * 16) * 40, 40, lane, 0);
#pragma unroll
    for (int i = 0; i < 2; ++i)
#pragma unroll
      for (int j = 0; j < 4; ++j)
        acc[i][j] = wmma_bf16(af[i], bfr[j], acc[i][j]);
    if (pre) stage_wait();
    __syncthreads();
    cur ^= 1;
  }

  const int half = lane >> 4, col = lane & 15;
#pragma unroll
  for (int i = 0; i < 2; ++i)
#pragma unroll
    for (int j = 0; j < 4; ++j)
#pragma unroll
      for (int r = 0; r < 8; ++r) {
        const int row = m0 + wm + i * 16 + half * 8 + r;
        const int cn  = n0 + wn + j * 16 + col;
        const float v = acc[i][j][r] + bias[cn];
        if (BF16OUT)
          ((bf16*)Cout)[(size_t)row * N + cn] = (bf16)v;
        else
          ((float*)Cout)[(size_t)row * N + cn] = v;
      }
}

// ---------------------------------------------------------------------------
// Flash-style GQA attention.
// grid = (S/64, H=32, B=4), block = 256 (8 waves). 64 queries per block.
// Q:  [B*S, 2048] bf16 (col = h*64+d)
// K:  [B*S, 512]  bf16 (col = kvh*64+d)  -> key-major tile == N-major for K^T
// Vt: per batch [512, 1024] bf16 (row = kvh*64+d, col = s) -> N-major for P@V
// ctx out: [B*S, 2048] bf16.
// ---------------------------------------------------------------------------
__global__ __launch_bounds__(256) void gqa_attention_kernel(
    const bf16* __restrict__ Q, const bf16* __restrict__ Kk,
    const bf16* __restrict__ Vt, bf16* __restrict__ Ctx) {
  const int S = 1024, DM = 2048, KVD = 512;
  __shared__ bf16 Qs[64][72];
  __shared__ bf16 Ks[64][72];    // [key][d]
  __shared__ bf16 Vts[64][72];   // [d][key]
  __shared__ bf16 Ps[64][72];    // [q][key]
  __shared__ float Ss[64][65];
  __shared__ float Pm[64][4], Psum[64][4];
  __shared__ float mrow[64], lrow[64], arow[64];

  const int tid = threadIdx.x, lane = tid & 31, wave = tid >> 5;
  const int half = lane >> 4, col = lane & 15;
  const int qt = blockIdx.x * 64;
  const int h  = blockIdx.y;
  const int b  = blockIdx.z;
  const int kvh = h >> 2;  // repeat_interleave: head h -> kv head h/4

  const size_t qbase = (size_t)b * S * DM + (size_t)h * 64;
  const size_t kbase = (size_t)b * S * KVD + (size_t)kvh * 64;
  const size_t vbase = ((size_t)b * KVD + (size_t)kvh * 64) * S;  // Vt origin

  // async-stage the 64x64 Q tile
#pragma unroll
  for (int c = 0; c < 2; ++c) {
    const int idx = tid + c * 256;
    const int r = idx >> 3, cq = (idx & 7) * 8;
    stage_b128(&Qs[r][cq], &Q[qbase + (size_t)(qt + r) * DM + cq]);
  }
  if (tid < 64) { mrow[tid] = -1e30f; lrow[tid] = 0.0f; }

  const int wm = (wave & 3) * 16;   // query-row tile
  const int wn = (wave >> 2) * 32;  // two 16-col tiles
  const int sr = tid >> 2, sseg = (tid & 3) * 16;  // softmax: 4 threads/row
  f32x8 oacc[2] = {};

  for (int kt = 0; kt < S; kt += 64) {
    // async-stage K chunk [key][d] and V^T chunk [d][key]
#pragma unroll
    for (int c = 0; c < 2; ++c) {
      const int idx = tid + c * 256;
      const int r = idx >> 3, cq = (idx & 7) * 8;
      stage_b128(&Ks[r][cq], &Kk[kbase + (size_t)(kt + r) * KVD + cq]);
      stage_b128(&Vts[r][cq], &Vt[vbase + (size_t)r * S + kt + cq]);
    }
    stage_wait();
    __syncthreads();

    // scores = Q @ K^T  (d=64 -> two k-steps of 32)
    f32x8 sacc[2] = {};
#pragma unroll
    for (int kk = 0; kk < 64; kk += 32) {
      bf16x16 a  = a_frag(&Qs[wm][0], 72, lane, kk);
      bf16x16 b0 = bn_frag(&Ks[wn][0], 72, lane, kk);
      bf16x16 b1 = bn_frag(&Ks[wn + 16][0], 72, lane, kk);
      sacc[0] = wmma_bf16(a, b0, sacc[0]);
      sacc[1] = wmma_bf16(a, b1, sacc[1]);
    }
#pragma unroll
    for (int j = 0; j < 2; ++j)
#pragma unroll
      for (int r = 0; r < 8; ++r)
        Ss[wm + half * 8 + r][wn + j * 16 + col] = sacc[j][r];
    __syncthreads();

    // phase 1 - partial max (4 threads per row)
    float pm = -1e30f;
    for (int c2 = 0; c2 < 16; ++c2) pm = fmaxf(pm, Ss[sr][sseg + c2]);
    Pm[sr][tid & 3] = pm;
    __syncthreads();

    // phase 2 - combine, update running max / alpha
    if (tid < 64) {
      const float m_old = mrow[tid];
      float mx = fmaxf(fmaxf(Pm[tid][0], Pm[tid][1]), fmaxf(Pm[tid][2], Pm[tid][3]));
      mx = fmaxf(mx * 0.125f, m_old);
      const float alpha = __expf(m_old - mx);
      mrow[tid] = mx;
      arow[tid] = alpha;
      lrow[tid] *= alpha;
    }
    __syncthreads();

    // phase 3 - probabilities (b128 DS stores) + partial sums
    {
      const float mx = mrow[sr];
      union { bf16 hh[16]; uint4 u[2]; } pk;
      float ls = 0.0f;
#pragma unroll
      for (int c2 = 0; c2 < 16; ++c2) {
        const float p = __expf(Ss[sr][sseg + c2] * 0.125f - mx);
        pk.hh[c2] = (bf16)p;
        ls += p;
      }
      *(uint4*)&Ps[sr][sseg]     = pk.u[0];
      *(uint4*)&Ps[sr][sseg + 8] = pk.u[1];
      Psum[sr][tid & 3] = ls;
    }
    __syncthreads();

    if (tid < 64)
      lrow[tid] += Psum[tid][0] + Psum[tid][1] + Psum[tid][2] + Psum[tid][3];

    // rescale running output, then O += P @ V  (keys=64 -> two k-steps)
#pragma unroll
    for (int j = 0; j < 2; ++j)
#pragma unroll
      for (int r = 0; r < 8; ++r) oacc[j][r] *= arow[wm + half * 8 + r];
#pragma unroll
    for (int kk = 0; kk < 64; kk += 32) {
      bf16x16 a  = a_frag(&Ps[wm][0], 72, lane, kk);
      bf16x16 b0 = bn_frag(&Vts[wn][0], 72, lane, kk);
      bf16x16 b1 = bn_frag(&Vts[wn + 16][0], 72, lane, kk);
      oacc[0] = wmma_bf16(a, b0, oacc[0]);
      oacc[1] = wmma_bf16(a, b1, oacc[1]);
    }
    __syncthreads();  // protect LDS tiles before next chunk
  }

  // normalize and store ctx (bf16, [B*S, 2048], col = h*64 + n)
#pragma unroll
  for (int j = 0; j < 2; ++j)
#pragma unroll
    for (int r = 0; r < 8; ++r) {
      const int row = wm + half * 8 + r;
      const float o = oacc[j][r] / lrow[row];
      Ctx[qbase + (size_t)(qt + row) * DM + wn + j * 16 + col] = (bf16)o;
    }
}

// ---------------------------------------------------------------------------
// Host-side orchestration
// ---------------------------------------------------------------------------
extern "C" void kernel_launch(void* const* d_in, const int* in_sizes, int n_in,
                              void* d_out, int out_size, void* d_ws, size_t ws_size,
                              hipStream_t stream) {
  const float* x  = (const float*)d_in[0];
  const float* Wq = (const float*)d_in[1];
  const float* bq = (const float*)d_in[2];
  const float* Wk = (const float*)d_in[3];
  const float* bk = (const float*)d_in[4];
  const float* Wv = (const float*)d_in[5];
  const float* bv = (const float*)d_in[6];
  const float* Wo = (const float*)d_in[7];
  const float* bo = (const float*)d_in[8];
  float* out = (float*)d_out;

  const int Mr = 4096;   // B*S
  const int DM = 2048;
  const int KVD = 512;
  const int S = 1024;

  bf16* p   = (bf16*)d_ws;
  bf16* xb  = p; p += (size_t)Mr * DM;
  bf16* Wqt = p; p += (size_t)DM * DM;    // [N=2048][K=2048]
  bf16* Wkt = p; p += (size_t)KVD * DM;   // [N=512][K=2048]
  bf16* Wvt = p; p += (size_t)KVD * DM;
  bf16* Wot = p; p += (size_t)DM * DM;
  bf16* Qb  = p; p += (size_t)Mr * DM;
  bf16* Kb  = p; p += (size_t)Mr * KVD;
  bf16* Vb  = p; p += (size_t)Mr * KVD;
  bf16* VtB = p; p += (size_t)Mr * KVD;   // per batch: [512][1024]
  bf16* Cxb = p; p += (size_t)Mr * DM;

  // activations: plain convert; weights: convert + transpose to N-major
  cvt_f32_bf16<<<dim3(Mr * DM / 1024), dim3(256), 0, stream>>>(x, xb, Mr * DM);
  cvt_t_f32_bf16<<<dim3(DM / 32, DM / 32), dim3(256), 0, stream>>>(Wq, Wqt, DM, DM);
  cvt_t_f32_bf16<<<dim3(KVD / 32, DM / 32), dim3(256), 0, stream>>>(Wk, Wkt, DM, KVD);
  cvt_t_f32_bf16<<<dim3(KVD / 32, DM / 32), dim3(256), 0, stream>>>(Wv, Wvt, DM, KVD);
  cvt_t_f32_bf16<<<dim3(DM / 32, DM / 32), dim3(256), 0, stream>>>(Wo, Wot, DM, DM);

  // projections (bf16 out)
  gemm_bias_kernel<true><<<dim3(Mr / 128, DM / 128), dim3(256), 0, stream>>>(
      xb, Wqt, bq, (void*)Qb, Mr, DM, DM);
  gemm_bias_kernel<true><<<dim3(Mr / 128, KVD / 128), dim3(256), 0, stream>>>(
      xb, Wkt, bk, (void*)Kb, Mr, KVD, DM);
  gemm_bias_kernel<true><<<dim3(Mr / 128, KVD / 128), dim3(256), 0, stream>>>(
      xb, Wvt, bv, (void*)Vb, Mr, KVD, DM);

  // V -> V^T per batch: [1024 x 512] -> [512 x 1024]
  transpose_bf16<<<dim3(KVD / 32, S / 32, 4), dim3(256), 0, stream>>>(Vb, VtB, S,
                                                                      KVD);

  // flash GQA attention
  gqa_attention_kernel<<<dim3(S / 64, 32, 4), dim3(256), 0, stream>>>(Qb, Kb, VtB,
                                                                      Cxb);

  // output projection, f32 straight into d_out
  gemm_bias_kernel<false><<<dim3(Mr / 128, DM / 128), dim3(256), 0, stream>>>(
      Cxb, Wot, bo, (void*)out, Mr, DM, DM);
}
